// SuperHeadAttention_16612933501329
// MI455X (gfx1250) — matrix operand order repaired and checked by reading, not compiled, not run on hardware
//
#include <hip/hip_runtime.h>
#include <hip/hip_bf16.h>
#include <math.h>

#define B_ 16
#define S_ 2048
#define H_ 1024
#define U_ 512
#define NH 8
#define TOPK 100

typedef __attribute__((ext_vector_type(16))) __bf16 v16bf;
typedef __attribute__((ext_vector_type(8)))  float  v8f;

__device__ __forceinline__ unsigned short f2bf(float f) {
    union { float f; unsigned int u; } x; x.f = f;
    unsigned int u = x.u;
    u += 0x7FFFu + ((u >> 16) & 1u);   // round-to-nearest-even
    return (unsigned short)(u >> 16);
}
__device__ __forceinline__ unsigned int pack2bf(float a, float b) {
    return (unsigned int)f2bf(a) | ((unsigned int)f2bf(b) << 16);
}

// ---------------------------------------------------------------------------
// Repack W1 [N,H,U] f32 -> bf16 tiles in WMMA-B lane layout:
// w1pack[((n*32+ct)*32+ks)*32+lane] = 16 bf16 (8 dwords):
//   column u = ct*16 + (lane&15); K = ks*32 + (lane>=16 ? 16 : 0) + {2v,2v+1}
// ---------------------------------------------------------------------------
__global__ void pack_w1(const float* __restrict__ W1, unsigned int* __restrict__ w1pack) {
    int t = blockIdx.x * blockDim.x + threadIdx.x;   // 8*32*32*32 = 262144
    int lane = t & 31;
    int ks   = (t >> 5) & 31;
    int ct   = (t >> 10) & 31;
    int n    = t >> 15;
    int u     = ct * 16 + (lane & 15);
    int kbase = ks * 32 + ((lane >= 16) ? 16 : 0);
    const float* w = W1 + (size_t)n * H_ * U_ + u;
    unsigned int* dst = w1pack + (size_t)t * 8;
#pragma unroll
    for (int v = 0; v < 8; ++v) {
        float a = w[(size_t)(kbase + 2 * v) * U_];
        float b = w[(size_t)(kbase + 2 * v + 1) * U_];
        dst[v] = pack2bf(a, b);
    }
}

// qbias[n,b,u] = b1[n,u] + b2[n,u] + sum_h query[b,h]*W2[n,h,u]
__global__ void qbias_kernel(const float* __restrict__ query, const float* __restrict__ W2,
                             const float* __restrict__ b1, const float* __restrict__ b2,
                             float* __restrict__ qbias) {
    int t = blockIdx.x * blockDim.x + threadIdx.x;   // 8*16*512 = 65536, u fastest
    int u = t & (U_ - 1);
    int b = (t >> 9) & 15;
    int n = t >> 13;
    const float* q = query + b * H_;
    const float* w = W2 + (size_t)n * H_ * U_ + u;
    float acc = b1[n * U_ + u] + b2[n * U_ + u];
    for (int h = 0; h < H_; ++h) acc += q[h] * w[(size_t)h * U_];
    qbias[t] = acc;
}

// VwW[n,u] = Ww[n] * Vw[n,u]
__global__ void vww_kernel(const float* __restrict__ Vw, const float* __restrict__ Ww,
                           float* __restrict__ VwW) {
    int t = blockIdx.x * blockDim.x + threadIdx.x;   // 4096
    VwW[t] = Vw[t] * Ww[t >> 9];
}

// ---------------------------------------------------------------------------
// Main WMMA kernel: 32 rows of (b,s) per workgroup (two 16-row M tiles sharing
// every B fetch), full K=1024, all 8 heads (one head-pair per wave).
// score[row] += sum_{n,u} Ww[n]*Vw[n,u]*tanh( (values@W1)[row,n,u] + qbias[n,b,u] )
// ---------------------------------------------------------------------------
__global__ void __launch_bounds__(128) score_gemm(
    const float* __restrict__ values, const unsigned int* __restrict__ w1pack,
    const float* __restrict__ qbias, const float* __restrict__ VwW,
    float* __restrict__ score)
{
    __shared__ unsigned int ldsA[32 * 2 * 32 * 8];   // [ks][tile][lane][8 dwords] = 64 KB
    const int tid = threadIdx.x;
    const int rowbase = blockIdx.x * 32;             // flat row = b*S + s
    const int bb = rowbase >> 11;                    // batch index (S=2048)

    // Stage A: 32 rows x 1024 K of values -> bf16, WMMA-A lane layout, 2 tiles.
    for (int m = 0; m < 32; ++m) {
        const int t  = m >> 4;                       // which 16-row tile
        const int ml = m & 15;
        const float* src = values + ((size_t)rowbase + m) * H_;
        for (int p = tid; p < 512; p += 128) {
            int k = p * 2;
            float a0 = src[k], a1 = src[k + 1];
            int ks = k >> 5;
            int kl = k & 31;
            int lane, v;
            lane = ml + ((kl >= 16) ? 16 : 0);
            v = (kl & 15) >> 1;
            ldsA[(((ks * 2 + t) * 32) + lane) * 8 + v] = pack2bf(a0, a1);
        }
    }
    __syncthreads();

    const int wave = tid >> 5;
    const int lane = tid & 31;
    const int col  = lane & 15;
    const int half = lane >> 4;

    float sacc0[8], sacc1[8];
#pragma unroll
    for (int r = 0; r < 8; ++r) { sacc0[r] = 0.f; sacc1[r] = 0.f; }

    for (int hh = 0; hh < 2; ++hh) {
        const int nh = wave * 2 + hh;                // this wave's head
        for (int ct = 0; ct < 32; ++ct) {            // 16-wide column tile within head
            v8f c0 = {}, c1 = {};
            const unsigned int* bp =
                w1pack + (((size_t)nh * 32 + ct) * 32 * 32 + lane) * 8;
#pragma unroll 4
            for (int ks = 0; ks < 32; ++ks) {        // K loop: 2 WMMAs share one B fetch
                v16bf bmat = *reinterpret_cast<const v16bf*>(bp + (size_t)ks * 256);
                v16bf a0 = *reinterpret_cast<const v16bf*>(&ldsA[((ks * 2 + 0) * 32 + lane) * 8]);
                v16bf a1 = *reinterpret_cast<const v16bf*>(&ldsA[((ks * 2 + 1) * 32 + lane) * 8]);
                c0 = __builtin_amdgcn_wmma_f32_16x16x32_bf16(
                         false, a0, false, bmat, (short)0, c0, false, false);
                c1 = __builtin_amdgcn_wmma_f32_16x16x32_bf16(
                         false, a1, false, bmat, (short)0, c1, false, false);
            }
            // Fused epilogue: tanh + Vw*Ww reduction over u (this lane's column).
            const int   u  = ct * 16 + col;
            const float qb = qbias[((size_t)nh * 16 + bb) * U_ + u];
            const float vw = VwW[nh * U_ + u];
#pragma unroll
            for (int r = 0; r < 8; ++r) {
                sacc0[r] += vw * tanhf(c0[r] + qb);
                sacc1[r] += vw * tanhf(c1[r] + qb);
            }
        }
    }
    // Reduce the 16 columns held across lanes of each half-wave.
#pragma unroll
    for (int r = 0; r < 8; ++r) {
        float v0 = sacc0[r], v1 = sacc1[r];
        v0 += __shfl_xor(v0, 1, 32);  v1 += __shfl_xor(v1, 1, 32);
        v0 += __shfl_xor(v0, 2, 32);  v1 += __shfl_xor(v1, 2, 32);
        v0 += __shfl_xor(v0, 4, 32);  v1 += __shfl_xor(v1, 4, 32);
        v0 += __shfl_xor(v0, 8, 32);  v1 += __shfl_xor(v1, 8, 32);
        sacc0[r] = v0; sacc1[r] = v1;
    }
    if (col == 0) {
#pragma unroll
        for (int r = 0; r < 8; ++r) {
            atomicAdd(&score[rowbase +      r + half * 8], sacc0[r]);
            atomicAdd(&score[rowbase + 16 + r + half * 8], sacc1[r]);
        }
    }
}

// ---------------------------------------------------------------------------
// Per-batch top-100 selection + mask + unnormalized sigmoid.
// ---------------------------------------------------------------------------
__global__ void __launch_bounds__(256) topk_mask(
    const float* __restrict__ score, const float* __restrict__ Ww,
    const float* __restrict__ Vb, const float* __restrict__ Wb,
    float* __restrict__ w_out, float* __restrict__ masked_out)
{
    __shared__ float vals[S_];
    __shared__ unsigned char keep[S_];
    __shared__ float rmaxv[256];
    __shared__ int   rmaxi[256];
    const int b = blockIdx.x;
    const int tid = threadIdx.x;
    float cterm = Wb[0];
    for (int n = 0; n < NH; ++n) cterm += Ww[n] * Vb[n];
    float orig[8];
    for (int i = 0; i < 8; ++i) {
        int s = tid + i * 256;
        float v = score[b * S_ + s] + cterm;
        orig[i] = v;
        vals[s] = v;
        keep[s] = 0;
    }
    __syncthreads();
    for (int it = 0; it < TOPK; ++it) {
        float mv = -INFINITY; int mi = 0;
        for (int i = 0; i < 8; ++i) {
            int s = tid + i * 256;
            float v = vals[s];
            if (v > mv) { mv = v; mi = s; }
        }
        rmaxv[tid] = mv; rmaxi[tid] = mi;
        __syncthreads();
        for (int off = 128; off > 0; off >>= 1) {
            if (tid < off && rmaxv[tid + off] > rmaxv[tid]) {
                rmaxv[tid] = rmaxv[tid + off];
                rmaxi[tid] = rmaxi[tid + off];
            }
            __syncthreads();
        }
        int win = rmaxi[0];
        if (tid == (win & 255)) { keep[win] = 1; vals[win] = -INFINITY; }
        __syncthreads();
    }
    for (int i = 0; i < 8; ++i) {
        int s = tid + i * 256;
        float m = keep[s] ? orig[i] : 0.0f;
        masked_out[b * S_ + s] = m;
        w_out[b * S_ + s] = 1.0f / (1.0f + expf(-m));   // sigmoid(0)=0.5 for non-topk
    }
}

// w[b,s] /= sum_b w[b,s]   (normalization over the BATCH dim, as in reference)
__global__ void normalize_w(float* __restrict__ w_out) {
    int s = blockIdx.x * blockDim.x + threadIdx.x;      // 2048
    float v[B_]; float sum = 0.f;
#pragma unroll
    for (int b = 0; b < B_; ++b) { v[b] = w_out[b * S_ + s]; sum += v[b]; }
    float inv = 1.0f / sum;
#pragma unroll
    for (int b = 0; b < B_; ++b) w_out[b * S_ + s] = v[b] * inv;
}

// context[b,h] = sum_s w[b,s] * values[b,s,h]
__global__ void __launch_bounds__(256) context_kernel(
    const float* __restrict__ values, const float* __restrict__ w,
    float* __restrict__ context)
{
    __shared__ float lw[256];
    const int b  = blockIdx.x >> 2;
    const int hc = blockIdx.x & 3;
    const int h  = hc * 256 + threadIdx.x;
    float acc = 0.f;
    for (int sc = 0; sc < 8; ++sc) {
        __syncthreads();
        lw[threadIdx.x] = w[b * S_ + sc * 256 + threadIdx.x];
        __syncthreads();
        const float* vp = values + ((size_t)b * S_ + sc * 256) * H_ + h;
        for (int i = 0; i < 256; ++i)
            acc += lw[i] * vp[(size_t)i * H_];
    }
    context[b * H_ + h] = acc;
}

extern "C" void kernel_launch(void* const* d_in, const int* in_sizes, int n_in,
                              void* d_out, int out_size, void* d_ws, size_t ws_size,
                              hipStream_t stream) {
    (void)in_sizes; (void)n_in; (void)out_size; (void)ws_size;
    const float* query  = (const float*)d_in[0];
    const float* values = (const float*)d_in[1];
    const float* W1     = (const float*)d_in[2];
    const float* b1     = (const float*)d_in[3];
    const float* W2     = (const float*)d_in[4];
    const float* b2     = (const float*)d_in[5];
    const float* Vw     = (const float*)d_in[6];
    const float* Vb     = (const float*)d_in[7];
    const float* Ww     = (const float*)d_in[8];
    const float* Wb     = (const float*)d_in[9];

    char* ws = (char*)d_ws;
    unsigned int* w1pack = (unsigned int*)ws;                              // 8 MB
    float* qbias = (float*)(ws + (8u << 20));                              // 256 KB
    float* VwW   = (float*)(ws + (8u << 20) + (256u << 10));               // 16 KB
    float* score = (float*)(ws + (8u << 20) + (272u << 10));               // 128 KB

    float* ctx_out    = (float*)d_out;          // [16,1024]
    float* w_out      = ctx_out + B_ * H_;      // [16,2048]
    float* masked_out = w_out + B_ * S_;        // [16,2048]

    hipMemsetAsync(score, 0, B_ * S_ * sizeof(float), stream);
    pack_w1<<<262144 / 256, 256, 0, stream>>>(W1, w1pack);
    qbias_kernel<<<65536 / 256, 256, 0, stream>>>(query, W2, b1, b2, qbias);
    vww_kernel<<<4096 / 256, 256, 0, stream>>>(Vw, Ww, VwW);
    score_gemm<<<(B_ * S_) / 32, 128, 0, stream>>>(values, w1pack, qbias, VwW, score);
    topk_mask<<<B_, 256, 0, stream>>>(score, Ww, Vb, Wb, w_out, masked_out);
    normalize_w<<<S_ / 256, 256, 0, stream>>>(w_out);
    context_kernel<<<B_ * 4, 256, 0, stream>>>(values, w_out, ctx_out);
}
